// GPKG_PREDICT_76562087018582
// MI455X (gfx1250) — matrix operand order repaired and checked by reading, not compiled, not run on hardware
//
#include <hip/hip_runtime.h>
#include <hip/hip_bf16.h>

typedef __attribute__((ext_vector_type(16))) _Float16 v16h;
typedef __attribute__((ext_vector_type(8)))  _Float16 v8h;
typedef __attribute__((ext_vector_type(8)))  float    v8f;

#define DIM    200
#define KPAD   224
#define NPAD   208          // N padded to multiple of 16 (zero rows 200..207)
#define LDA    232          // LDS row stride (halves), 16B aligned
#define NENT   50000
#define NRELT  400          // 2*NUM_REL
#define EHALF  250000
#define ETOT   500000
#define BATCH  1024
#define NFILT  96
#define FLATSZ 38400

union V16H { v16h v; v8h h[2]; };

__device__ inline v8f vzero8f() {
  v8f z;
#pragma unroll
  for (int i = 0; i < 8; i++) z[i] = 0.f;
  return z;
}

// A fragment: lane l -> row m=l&15, group g=l>>4.
// elements 0..7  : k = 32*kstep + 8*g + i
// elements 8..15 : k = 32*kstep + 16 + 8*g + (i-8)
__device__ inline v16h load_a16(const _Float16* tile, long stride, int kstep) {
  int l = threadIdx.x & 31;
  int m = l & 15, g = l >> 4;
  const _Float16* p = tile + (long)m * stride + kstep * 32 + g * 8;
  V16H u;
  u.h[0] = *(const v8h*)(p);
  u.h[1] = *(const v8h*)(p + 16);
  return u.v;
}

// B fragment from Bt[n][k] (row-major, ldk halves per row, N pre-padded to tile
// multiple with zero rows -> no guard, no EXEC juggling in the inner loop):
// lane l -> col n=n0+(l&15), elements i<8 -> k = 32*kstep + 8*(l>>4) + i, i>=8 -> k = 32*kstep + 16 + 8*(l>>4) + (i-8)
__device__ inline v16h load_bt(const _Float16* Bt, long ldk, int n0, int kstep) {
  int l = threadIdx.x & 31;
  int n = n0 + (l & 15), g = l >> 4;
  const _Float16* p = Bt + (long)n * ldk + kstep * 32 + g * 8;
  V16H u;
  u.h[0] = *(const v8h*)(p);
  u.h[1] = *(const v8h*)(p + 16);
  return u.v;
}

__device__ inline v8f wmma16(v16h a, v16h b, v8f c) {
  return __builtin_amdgcn_wmma_f32_16x16x32_f16(false, a, false, b, (short)0, c, false, false);
}

// ---------------- utility kernels ----------------
__global__ void k_zero(float* p, long n) {
  long i = (long)blockIdx.x * 256 + threadIdx.x;
  if (i < n) p[i] = 0.f;
}

// transpose + f16-convert the four 200x200 weights into [208][224] Bt layout (zero padded)
__global__ void k_prepw(const float* w0, const float* w1, const float* w2, const float* w3,
                        _Float16* t0, _Float16* t1, _Float16* t2, _Float16* t3) {
  int i = blockIdx.x * 256 + threadIdx.x;
  if (i >= 4 * NPAD * KPAD) return;
  int which = i / (NPAD * KPAD), rem = i % (NPAD * KPAD);
  int n = rem / KPAD, k = rem % KPAD;
  const float* W = (which == 0) ? w0 : (which == 1) ? w1 : (which == 2) ? w2 : w3;
  _Float16* T   = (which == 0) ? t0 : (which == 1) ? t1 : (which == 2) ? t2 : t3;
  T[(long)n * KPAD + k] = (_Float16)((n < DIM && k < DIM) ? W[(long)k * DIM + n] : 0.f);
}

// fc_w rows are already K-contiguous; pad N 200->208 with zero rows
__global__ void k_prepfcw(const float* fcw, _Float16* fcw_h) {
  long i = (long)blockIdx.x * 256 + threadIdx.x;
  if (i >= (long)NPAD * FLATSZ) return;
  int n = (int)(i / FLATSZ);
  fcw_h[i] = (_Float16)((n < DIM) ? fcw[i] : 0.f);
}

__global__ void k_deg(const int* dst, float* deg) {
  int e = blockIdx.x * 256 + threadIdx.x;
  if (e < ETOT) atomicAdd(&deg[dst[e]], 1.f);
}

__global__ void k_dinv(float* deg) {
  int i = blockIdx.x * 256 + threadIdx.x;
  if (i < NENT) { float d = deg[i]; deg[i] = (d > 0.f) ? rsqrtf(fmaxf(d, 1.f)) : 0.f; }
}

// ---------------- edge GEMM + normalized scatter (WMMA) ----------------
__global__ __launch_bounds__(416) void k_edge(const float* emb, const float* rel,
                                              const int* src, const int* dst, const int* etype,
                                              const float* dinv, const _Float16* w_in_t,
                                              const _Float16* w_out_t, float* agg) {
  __shared__ _Float16 At[16 * LDA];
  __shared__ int   dst_s[16];
  __shared__ float nrm_s[16];
  int tid = threadIdx.x;
  int e0 = blockIdx.x * 16;
  for (int idx = tid; idx < 16 * KPAD; idx += 416) {
    int row = idx / KPAD, k = idx % KPAD;
    int e = e0 + row;
    float v = 0.f;
    if (k < DIM) v = emb[(long)src[e] * DIM + k] - rel[(long)etype[e] * DIM + k];
    At[row * LDA + k] = (_Float16)v;
  }
  if (tid < 16) {
    int e = e0 + tid;
    dst_s[tid] = dst[e];
    nrm_s[tid] = dinv[src[e]] * dinv[dst[e]];
  }
  __syncthreads();
  int n0 = (tid >> 5) * 16;
  const _Float16* Bt = (e0 < EHALF) ? w_in_t : w_out_t;
  v8f c = vzero8f();
#pragma unroll
  for (int ks = 0; ks < 7; ks++)
    c = wmma16(load_a16(At, LDA, ks), load_bt(Bt, KPAD, n0, ks), c);
  int l = tid & 31;
  int n = n0 + (l & 15), g = l >> 4;
  if (n < DIM) {
#pragma unroll
    for (int v = 0; v < 8; v++) {
      int m = v + 8 * g;
      atomicAdd(&agg[(long)dst_s[m] * DIM + n], c[v] * nrm_s[m]);
    }
  }
}

// ---------------- loop GEMM + tanh fuse (WMMA) ----------------
__global__ __launch_bounds__(416) void k_loop(const float* emb, const float* loop_rel,
                                              const _Float16* w_loop_t, const float* agg,
                                              float* all_ent, _Float16* ent_h) {
  __shared__ _Float16 At[16 * LDA];
  int tid = threadIdx.x;
  int m0 = blockIdx.x * 16;
  for (int idx = tid; idx < 16 * KPAD; idx += 416) {
    int row = idx / KPAD, k = idx % KPAD;
    float v = 0.f;
    if (k < DIM) v = emb[(long)(m0 + row) * DIM + k] - loop_rel[k];
    At[row * LDA + k] = (_Float16)v;
  }
  __syncthreads();
  int n0 = (tid >> 5) * 16;
  v8f c = vzero8f();
#pragma unroll
  for (int ks = 0; ks < 7; ks++)
    c = wmma16(load_a16(At, LDA, ks), load_bt(w_loop_t, KPAD, n0, ks), c);
  int l = tid & 31;
  int n = n0 + (l & 15), g = l >> 4;
  if (n < DIM) {
#pragma unroll
    for (int v = 0; v < 8; v++) {
      int mg = m0 + v + 8 * g;
      float val = tanhf((agg[(long)mg * DIM + n] + c[v]) * (1.f / 3.f));
      all_ent[(long)mg * DIM + n] = val;
      ent_h[(long)mg * KPAD + n] = (_Float16)val;
    }
  }
}

// ---------------- relation GEMM (WMMA) ----------------
__global__ __launch_bounds__(416) void k_rel(const float* init_rel, const _Float16* w_rel_t,
                                             float* r_buf) {
  __shared__ _Float16 At[16 * LDA];
  int tid = threadIdx.x;
  int m0 = blockIdx.x * 16;
  for (int idx = tid; idx < 16 * KPAD; idx += 416) {
    int row = idx / KPAD, k = idx % KPAD;
    At[row * LDA + k] = (_Float16)((k < DIM) ? init_rel[(long)(m0 + row) * DIM + k] : 0.f);
  }
  __syncthreads();
  int n0 = (tid >> 5) * 16;
  v8f c = vzero8f();
#pragma unroll
  for (int ks = 0; ks < 7; ks++)
    c = wmma16(load_a16(At, LDA, ks), load_bt(w_rel_t, KPAD, n0, ks), c);
  int l = tid & 31;
  int n = n0 + (l & 15), g = l >> 4;
  if (n < DIM) {
#pragma unroll
    for (int v = 0; v < 8; v++) r_buf[(long)(m0 + v + 8 * g) * DIM + n] = c[v];
  }
}

// ---------------- chequer permute + BN0 stats ----------------
__global__ void k_cheq(const float* all_ent, const float* r_buf, const int* sub, const int* rel,
                       const int* perm, float* x0, float* bn0s) {
  __shared__ float s2[2];
  int tid = threadIdx.x;
  if (tid < 2) s2[tid] = 0.f;
  __syncthreads();
  int i = blockIdx.x * 256 + tid;
  float v = 0.f;
  if (i < BATCH * 400) {
    int b = i / 400, j = i % 400;
    int p = perm[j];
    v = (p < DIM) ? all_ent[(long)sub[b] * DIM + p] : r_buf[(long)rel[b] * DIM + (p - DIM)];
    x0[i] = v;
  }
  atomicAdd(&s2[0], v);
  atomicAdd(&s2[1], v * v);
  __syncthreads();
  if (tid < 2) atomicAdd(&bn0s[tid], s2[tid]);
}

// ---------------- BN0 apply + circular pad ----------------
__global__ void k_pad(const float* x0, const float* bn0s, const float* g0, const float* b0,
                      float* xp) {
  int i = blockIdx.x * 256 + threadIdx.x;
  if (i >= BATCH * 784) return;
  int b = i / 784, rem = i % 784;
  int y = rem / 28, x = rem % 28;
  int sy = (y + 16) % 20, sx = (x + 16) % 20;
  float v = x0[b * 400 + sy * 20 + sx];
  const float N = (float)(BATCH * 400);
  float mean = bn0s[0] / N;
  float var  = bn0s[1] / N - mean * mean;
  xp[i] = (v - mean) * rsqrtf(var + 1e-5f) * g0[0] + b0[0];
}

// ---------------- direct 9x9 conv + BN1 stats ----------------
__global__ __launch_bounds__(256) void k_conv(const float* xp, const float* filt,
                                              _Float16* conv_h, float* bn1_sum, float* bn1_sq) {
  __shared__ float tile[784];
  __shared__ float fs[NFILT * 81];
  __shared__ float cs[NFILT], cq[NFILT];
  int tid = threadIdx.x, b = blockIdx.x;
  for (int i = tid; i < 784; i += 256) tile[i] = xp[b * 784 + i];
  for (int i = tid; i < NFILT * 81; i += 256) fs[i] = filt[i];
  if (tid < NFILT) { cs[tid] = 0.f; cq[tid] = 0.f; }
  __syncthreads();
  for (int o = tid; o < FLATSZ; o += 256) {
    int f = o / 400, yx = o % 400;
    int y = yx / 20, x = yx % 20;
    const float* fp = fs + f * 81;
    float acc = 0.f;
#pragma unroll
    for (int dy = 0; dy < 9; dy++)
#pragma unroll
      for (int dx = 0; dx < 9; dx++)
        acc += tile[(y + dy) * 28 + (x + dx)] * fp[dy * 9 + dx];
    conv_h[(long)b * FLATSZ + o] = (_Float16)acc;
    atomicAdd(&cs[f], acc);
    atomicAdd(&cq[f], acc * acc);
  }
  __syncthreads();
  if (tid < NFILT) { atomicAdd(&bn1_sum[tid], cs[tid]); atomicAdd(&bn1_sq[tid], cq[tid]); }
}

__global__ void k_bn1fin(const float* s, const float* q, const float* g, const float* bt,
                         float* a1, float* a0) {
  int f = threadIdx.x;
  if (f < NFILT) {
    const float N = (float)(BATCH * 400);
    float mean = s[f] / N, var = q[f] / N - mean * mean;
    float sc = rsqrtf(var + 1e-5f) * g[f];
    a1[f] = sc;
    a0[f] = bt[f] - mean * sc;
  }
}

__global__ void k_bn1apply(_Float16* conv_h, const float* a1, const float* a0) {
  long i = (long)blockIdx.x * 256 + threadIdx.x;
  if (i >= (long)BATCH * FLATSZ) return;
  int ch = (int)(i % FLATSZ) / 400;
  float v = (float)conv_h[i] * a1[ch] + a0[ch];
  conv_h[i] = (_Float16)fmaxf(v, 0.f);
}

// ---------------- FC GEMM (WMMA) + BN2 stats ----------------
__global__ __launch_bounds__(256) void k_fc(const _Float16* Ah, const _Float16* fcw_h,
                                            const float* fc_b, float* fc_out,
                                            float* bn2_sum, float* bn2_sq) {
  int wg = blockIdx.x * 8 + (threadIdx.x >> 5);
  if (wg >= 64 * 13) return;
  int mtile = wg / 13, nt = wg % 13;
  const _Float16* At = Ah + (long)mtile * 16 * FLATSZ;
  v8f c = vzero8f();
  for (int ks = 0; ks < FLATSZ / 32; ks++)
    c = wmma16(load_a16(At, FLATSZ, ks), load_bt(fcw_h, FLATSZ, nt * 16, ks), c);
  int l = threadIdx.x & 31;
  int n = nt * 16 + (l & 15), g = l >> 4;
  if (n < DIM) {
    float s = 0.f, q = 0.f;
    float bias = fc_b[n];
#pragma unroll
    for (int v = 0; v < 8; v++) {
      int b = mtile * 16 + v + 8 * g;
      float val = c[v] + bias;
      fc_out[(long)b * DIM + n] = val;
      s += val;
      q += val * val;
    }
    atomicAdd(&bn2_sum[n], s);
    atomicAdd(&bn2_sq[n], q);
  }
}

// ---------------- BN2 apply + ReLU -> f16 A matrix for logits ----------------
__global__ void k_bn2(const float* fc_out, const float* s, const float* q,
                      const float* g2, const float* b2, _Float16* x2) {
  int i = blockIdx.x * 256 + threadIdx.x;
  if (i >= BATCH * KPAD) return;
  int b = i / KPAD, k = i % KPAD;
  float y = 0.f;
  if (k < DIM) {
    float mean = s[k] / (float)BATCH, var = q[k] / (float)BATCH - mean * mean;
    float v = (fc_out[(long)b * DIM + k] - mean) * rsqrtf(var + 1e-5f) * g2[k] + b2[k];
    y = fmaxf(v, 0.f);
  }
  x2[i] = (_Float16)y;
}

// ---------------- logits GEMM (WMMA) + bias + sigmoid ----------------
__global__ __launch_bounds__(256) void k_logits(const _Float16* x2, const _Float16* ent_h,
                                                const float* bias, float* out) {
  int wg = blockIdx.x * 8 + (threadIdx.x >> 5);
  if (wg >= 64 * (NENT / 16)) return;
  int mtile = wg / (NENT / 16), nt = wg % (NENT / 16);
  const _Float16* At = x2 + (long)mtile * 16 * KPAD;
  v8f c = vzero8f();
#pragma unroll
  for (int ks = 0; ks < 7; ks++)
    c = wmma16(load_a16(At, KPAD, ks), load_bt(ent_h, KPAD, nt * 16, ks), c);
  int l = threadIdx.x & 31;
  int e = nt * 16 + (l & 15), g = l >> 4;
  float bi = bias[e];
#pragma unroll
  for (int v = 0; v < 8; v++) {
    int b = mtile * 16 + v + 8 * g;
    float val = c[v] + bi;
    out[(long)b * NENT + e] = 1.f / (1.f + __expf(-val));
  }
}

// ---------------- host launcher ----------------
extern "C" void kernel_launch(void* const* d_in, const int* in_sizes, int n_in,
                              void* d_out, int out_size, void* d_ws, size_t ws_size,
                              hipStream_t stream) {
  (void)in_sizes; (void)n_in; (void)out_size; (void)ws_size;
  const int*   sub       = (const int*)d_in[0];
  const int*   rel       = (const int*)d_in[1];
  // d_in[2] = neg_ents (unused in forward)
  const int*   ei        = (const int*)d_in[3];      // [2, ETOT]
  const int*   etype     = (const int*)d_in[4];
  const int*   perm      = (const int*)d_in[5];      // [1, 400]
  const float* init_emb  = (const float*)d_in[6];
  const float* init_rel  = (const float*)d_in[7];
  const float* w_in      = (const float*)d_in[8];
  const float* w_out     = (const float*)d_in[9];
  const float* w_loop    = (const float*)d_in[10];
  const float* w_rel     = (const float*)d_in[11];
  const float* loop_rel  = (const float*)d_in[12];
  const float* bn0_g     = (const float*)d_in[13];
  const float* bn0_b     = (const float*)d_in[14];
  const float* bn1_g     = (const float*)d_in[15];
  const float* bn1_b     = (const float*)d_in[16];
  const float* bn2_g     = (const float*)d_in[17];
  const float* bn2_b     = (const float*)d_in[18];
  const float* conv_filt = (const float*)d_in[19];
  const float* fc_w      = (const float*)d_in[20];
  const float* fc_b      = (const float*)d_in[21];
  const float* bias_ent  = (const float*)d_in[22];
  float* out = (float*)d_out;

  const int* src = ei;
  const int* dst = ei + ETOT;

  char* w = (char*)d_ws;
  size_t off = 0;
  auto alloc = [&](size_t bytes) -> void* {
    void* p = w + off;
    off = (off + bytes + 255) & ~(size_t)255;
    return p;
  };
  float*    deg      = (float*)alloc((size_t)NENT * 4);            // -> dinv in place
  float*    agg      = (float*)alloc((size_t)NENT * DIM * 4);
  float*    all_ent  = (float*)alloc((size_t)NENT * DIM * 4);
  _Float16* ent_h    = (_Float16*)alloc((size_t)NENT * KPAD * 2);
  _Float16* w_in_t   = (_Float16*)alloc((size_t)NPAD * KPAD * 2);
  _Float16* w_out_t  = (_Float16*)alloc((size_t)NPAD * KPAD * 2);
  _Float16* w_loop_t = (_Float16*)alloc((size_t)NPAD * KPAD * 2);
  _Float16* w_rel_t  = (_Float16*)alloc((size_t)NPAD * KPAD * 2);
  _Float16* fcw_h    = (_Float16*)alloc((size_t)NPAD * FLATSZ * 2);
  float*    r_buf    = (float*)alloc((size_t)NRELT * DIM * 4);
  float*    x0       = (float*)alloc((size_t)BATCH * 400 * 4);
  float*    xp       = (float*)alloc((size_t)BATCH * 784 * 4);
  _Float16* conv_h   = (_Float16*)alloc((size_t)BATCH * FLATSZ * 2);
  float*    fc_out   = (float*)alloc((size_t)BATCH * DIM * 4);
  _Float16* x2       = (_Float16*)alloc((size_t)BATCH * KPAD * 2);
  float*    stats    = (float*)alloc(786 * 4);
  float* bn0s    = stats;        // 2
  float* bn1_sum = stats + 2;    // 96
  float* bn1_sq  = stats + 98;   // 96
  float* bn1_a1  = stats + 194;  // 96
  float* bn1_a0  = stats + 290;  // 96
  float* bn2_sum = stats + 386;  // 200
  float* bn2_sq  = stats + 586;  // 200

  // init
  k_zero<<<(NENT + 255) / 256, 256, 0, stream>>>(deg, NENT);
  k_zero<<<((long)NENT * DIM + 255) / 256, 256, 0, stream>>>(agg, (long)NENT * DIM);
  k_zero<<<4, 256, 0, stream>>>(stats, 786);
  k_zero<<<((long)NENT * KPAD / 2 + 255) / 256, 256, 0, stream>>>((float*)ent_h, (long)NENT * KPAD / 2);
  // weight conversion
  k_prepw<<<(4 * NPAD * KPAD + 255) / 256, 256, 0, stream>>>(w_in, w_out, w_loop, w_rel,
                                                             w_in_t, w_out_t, w_loop_t, w_rel_t);
  k_prepfcw<<<((long)NPAD * FLATSZ + 255) / 256, 256, 0, stream>>>(fc_w, fcw_h);
  // degree normalization
  k_deg<<<(ETOT + 255) / 256, 256, 0, stream>>>(dst, deg);
  k_dinv<<<(NENT + 255) / 256, 256, 0, stream>>>(deg);
  // encoder GEMMs
  k_edge<<<ETOT / 16, 416, 0, stream>>>(init_emb, init_rel, src, dst, etype, deg,
                                        w_in_t, w_out_t, agg);
  k_loop<<<NENT / 16, 416, 0, stream>>>(init_emb, loop_rel, w_loop_t, agg, all_ent, ent_h);
  k_rel<<<NRELT / 16, 416, 0, stream>>>(init_rel, w_rel_t, r_buf);
  // decoder
  k_cheq<<<(BATCH * 400 + 255) / 256, 256, 0, stream>>>(all_ent, r_buf, sub, rel, perm, x0, bn0s);
  k_pad<<<(BATCH * 784 + 255) / 256, 256, 0, stream>>>(x0, bn0s, bn0_g, bn0_b, xp);
  k_conv<<<BATCH, 256, 0, stream>>>(xp, conv_filt, conv_h, bn1_sum, bn1_sq);
  k_bn1fin<<<1, 128, 0, stream>>>(bn1_sum, bn1_sq, bn1_g, bn1_b, bn1_a1, bn1_a0);
  k_bn1apply<<<((long)BATCH * FLATSZ + 255) / 256, 256, 0, stream>>>(conv_h, bn1_a1, bn1_a0);
  k_fc<<<(64 * 13 + 7) / 8, 256, 0, stream>>>(conv_h, fcw_h, fc_b, fc_out, bn2_sum, bn2_sq);
  k_bn2<<<(BATCH * KPAD + 255) / 256, 256, 0, stream>>>(fc_out, bn2_sum, bn2_sq, bn2_g, bn2_b, x2);
  k_logits<<<64 * (NENT / 16) / 8, 256, 0, stream>>>(x2, ent_h, bias_ent, out);
}
